// AtlasAttentionWrapper_59571196395971
// MI455X (gfx1250) — matrix operand order repaired and checked by reading, not compiled, not run on hardware
//
#include <hip/hip_runtime.h>
#include <hip/hip_bf16.h>

typedef __attribute__((ext_vector_type(16))) __bf16 v16bf;
typedef __attribute__((ext_vector_type(8)))  __bf16 v8bf;
typedef __attribute__((ext_vector_type(4)))  __bf16 v4bf;
typedef __attribute__((ext_vector_type(4)))  float  v4f;
typedef __attribute__((ext_vector_type(8)))  float  v8f;

#define LOG_THETA 9.2103403719761836f  // ln(10000)

// gfx1250 async global->LDS path via inline asm (bypasses toolchain-specific
// builtin signatures; see CDNA5 ISA).
#if defined(__gfx1250__)
#define USE_ASYNC_LDS 1
#else
#define USE_ASYNC_LDS 0
#endif

__device__ __forceinline__ v8f wmma_bf16(const v16bf& a, const v16bf& b, const v8f& c) {
  // D = A(16x32 bf16) * B(32x16 bf16) + C(16x16 f32)
  return __builtin_amdgcn_wmma_f32_16x16x32_bf16(false, a, false, b, (short)0, c, false, false);
}

// Packed 4-element load with fp32 -> bf16 conversion (or straight bf16 copy).
__device__ __forceinline__ v4bf load4_bf(const float* p) {
  v4f f = *(const v4f*)p;
  v4bf r;
#pragma unroll
  for (int i = 0; i < 4; ++i) r[i] = (__bf16)f[i];
  return r;
}
__device__ __forceinline__ v4bf load4_bf(const __bf16* p) { return *(const v4bf*)p; }

#if USE_ASYNC_LDS
// 16B global -> LDS async copy: GLOBAL_LOAD_ASYNC_TO_LDS_B128 (ASYNCcnt).
__device__ __forceinline__ void async_copy16(void* lds, const void* g) {
  __attribute__((address_space(3))) void* lp =
      (__attribute__((address_space(3))) void*)lds;   // 32-bit LDS address
  asm volatile("global_load_async_to_lds_b128 %0, %1, off"
               :
               : "v"(lp), "v"(g)
               : "memory");
}
__device__ __forceinline__ void wait_async0() {
  asm volatile("s_wait_asynccnt 0x0" ::: "memory");
}
#endif

// ---------------------------------------------------------------------------
// GEMM: C[m][n] = sum_k A[m][k] * B[n][k]   (nn.Linear convention, B = weight)
// A: TA (float or __bf16), B: fp32, C: TC (__bf16 or float)
// Tiles: 128x128 block, BK=32, 8 waves, wave = 2(M) x 4(N) subtiles of 16x16.
// ---------------------------------------------------------------------------
template<typename TA, typename TC>
__global__ __launch_bounds__(256)
void gemm_wmma(const TA* __restrict__ A, const float* __restrict__ B,
               TC* __restrict__ C, int M, int N, int K)
{
  constexpr int BM = 128, BN = 128, BK = 32, LDT = 40; // stride 40 -> 80B rows, 16B aligned
  __shared__ __bf16 As[BM][LDT];
  __shared__ __bf16 Bs[BN][LDT];

  const int tid  = threadIdx.x;
  const int lane = tid & 31;
  const int wave = tid >> 5;       // 0..7
  const int wm   = wave >> 1;      // 0..3 -> rows wm*32
  const int wn   = wave & 1;       // 0..1 -> cols wn*64
  const int m0   = blockIdx.y * BM;
  const int n0   = blockIdx.x * BN;

  v8f acc[2][4];
#pragma unroll
  for (int mi = 0; mi < 2; ++mi)
#pragma unroll
    for (int ni = 0; ni < 4; ++ni)
#pragma unroll
      for (int g = 0; g < 8; ++g) acc[mi][ni][g] = 0.0f;

  const int lr = tid >> 3;         // 0..31 staging row
  const int lc = (tid & 7) * 4;    // staging col group

  const int cl = lane & 15;
  const int hl = (lane < 16) ? 0 : 8;      // A-frag K-half
  const int bk = (lane >> 4) * 8;          // B-frag K-half

  for (int k0 = 0; k0 < K; k0 += BK) {
    // ---- stage A and B tiles into LDS as bf16 (packed b64 stores) ----
#pragma unroll
    for (int it = 0; it < 4; ++it) {
      const int row = lr + it * 32;
      *(v4bf*)&As[row][lc] = load4_bf(A + (size_t)(m0 + row) * K + k0 + lc);
      *(v4bf*)&Bs[row][lc] = load4_bf(B + (size_t)(n0 + row) * K + k0 + lc);
    }
    if (k0 + BK < K) {  // prefetch next tiles -> global_prefetch_b8
      __builtin_prefetch((const void*)(B + (size_t)(n0 + lr) * K + k0 + BK + lc), 0, 1);
      __builtin_prefetch((const void*)(A + (size_t)(m0 + lr) * K + k0 + BK + lc), 0, 1);
    }
    __syncthreads();

    // ---- build fragments ----
    v16bf af[2], bf[4];
#pragma unroll
    for (int mi = 0; mi < 2; ++mi) {
      const int r = wm * 32 + mi * 16 + cl;
      v8bf lo = *(const v8bf*)&As[r][hl];
      v8bf hi = *(const v8bf*)&As[r][hl + 16];
#pragma unroll
      for (int i = 0; i < 8; ++i) { af[mi][i] = lo[i]; af[mi][i + 8] = hi[i]; }
    }
#pragma unroll
    for (int ni = 0; ni < 4; ++ni) {
      const int c = wn * 64 + ni * 16 + cl;
      v8bf lo = *(const v8bf*)&Bs[c][bk];
      v8bf hi = *(const v8bf*)&Bs[c][bk + 16];
#pragma unroll
      for (int i = 0; i < 8; ++i) { bf[ni][i] = lo[i]; bf[ni][i + 8] = hi[i]; }
    }

#pragma unroll
    for (int mi = 0; mi < 2; ++mi)
#pragma unroll
      for (int ni = 0; ni < 4; ++ni)
        acc[mi][ni] = wmma_bf16(af[mi], bf[ni], acc[mi][ni]);

    __syncthreads();
  }

  // ---- store C (C layout: VGPR g -> row g / g+8 per lane half, col = lane%16)
  const int rh = (lane >> 4) * 8;
#pragma unroll
  for (int mi = 0; mi < 2; ++mi)
#pragma unroll
    for (int ni = 0; ni < 4; ++ni)
#pragma unroll
      for (int g = 0; g < 8; ++g) {
        const int r = m0 + wm * 32 + mi * 16 + g + rh;
        const int c = n0 + wn * 64 + ni * 16 + cl;
        C[(size_t)r * N + c] = (TC)acc[mi][ni][g];
      }
}

// ---------------------------------------------------------------------------
// Past KV ingest: fp32 -> bf16 cache, RoPE on keys (positions 0..1023).
// grid (1024 pos, 8 kv-heads), 64 threads (one freq pair each)
// ---------------------------------------------------------------------------
__global__ __launch_bounds__(64)
void ingest_past_kernel(const float* __restrict__ pk, const float* __restrict__ pv,
                        __bf16* __restrict__ Kc, __bf16* __restrict__ Vc)
{
  const int pos = blockIdx.x, kv = blockIdx.y, t = threadIdx.x;
  const float* k = pk + ((size_t)kv * 1024 + pos) * 128;
  const float* v = pv + ((size_t)kv * 1024 + pos) * 128;
  __bf16* ko = Kc + ((size_t)kv * 2048 + pos) * 128;
  __bf16* vo = Vc + ((size_t)kv * 2048 + pos) * 128;
  const float inv = __expf(-LOG_THETA * ((float)t / 64.0f));
  float sn, cs; __sincosf((float)pos * inv, &sn, &cs);
  const float a = k[t], b = k[t + 64];
  ko[t]      = (__bf16)(a * cs - b * sn);
  ko[t + 64] = (__bf16)(b * cs + a * sn);
  vo[t]      = (__bf16)v[t];
  vo[t + 64] = (__bf16)v[t + 64];
}

// RoPE on fresh queries in-place (positions 1024+s). grid (1024, 32), 64 thr.
__global__ __launch_bounds__(64)
void rope_q_kernel(__bf16* __restrict__ Q)
{
  const int s = blockIdx.x, h = blockIdx.y, t = threadIdx.x;
  const int pos = 1024 + s;
  __bf16* q = Q + (size_t)s * 4096 + h * 128;
  const float inv = __expf(-LOG_THETA * ((float)t / 64.0f));
  float sn, cs; __sincosf((float)pos * inv, &sn, &cs);
  const float a = (float)q[t], b = (float)q[t + 64];
  q[t]      = (__bf16)(a * cs - b * sn);
  q[t + 64] = (__bf16)(b * cs + a * sn);
}

// RoPE + scatter fresh K, scatter fresh V into caches. grid (1024, 8), 64 thr.
__global__ __launch_bounds__(64)
void scatter_new_kv_kernel(const __bf16* __restrict__ Kp, const __bf16* __restrict__ Vp,
                           __bf16* __restrict__ Kc, __bf16* __restrict__ Vc)
{
  const int s = blockIdx.x, kv = blockIdx.y, t = threadIdx.x;
  const int pos = 1024 + s;
  const __bf16* kin = Kp + (size_t)s * 1024 + kv * 128;
  const __bf16* vin = Vp + (size_t)s * 1024 + kv * 128;
  __bf16* ko = Kc + ((size_t)kv * 2048 + pos) * 128;
  __bf16* vo = Vc + ((size_t)kv * 2048 + pos) * 128;
  const float inv = __expf(-LOG_THETA * ((float)t / 64.0f));
  float sn, cs; __sincosf((float)pos * inv, &sn, &cs);
  const float a = (float)kin[t], b = (float)kin[t + 64];
  ko[t]      = (__bf16)(a * cs - b * sn);
  ko[t + 64] = (__bf16)(b * cs + a * sn);
  vo[t]      = vin[t];
  vo[t + 64] = vin[t + 64];
}

// ---------------------------------------------------------------------------
// Flash attention. One block = 64 q rows x one head. 4 waves; wave owns a
// 16-row band. KV walked in chunks of 64, online softmax in fp32.
// grid (16 q-tiles, 32 heads), 128 threads.
// ---------------------------------------------------------------------------
__global__ __launch_bounds__(128)
void flash_attn_kernel(const __bf16* __restrict__ Q,   // [1024][4096]
                       const __bf16* __restrict__ Kc,  // [8][2048][128]
                       const __bf16* __restrict__ Vc,  // [8][2048][128]
                       __bf16* __restrict__ Oa)        // [1024][4096]
{
  constexpr int DQ = 136, DV = 72;
  __shared__ __bf16 Qs[64][DQ];
  __shared__ __bf16 Ks[64][DQ];
  __shared__ __bf16 Vt[128][DV];        // V transposed: [d][k]
  __shared__ __bf16 Ps[4][16][DV];      // per-wave P strip

  const int tid = threadIdx.x, lane = tid & 31, w = tid >> 5;
  const int qt = blockIdx.x, h = blockIdx.y;
  const int kvh = h >> 2;               // GQA: 4 query heads per kv head
  const int q0 = qt * 64;
  const float scale = 0.08838834764831845f;   // 1/sqrt(128)

  const int cl = lane & 15;
  const int hl = (lane < 16) ? 0 : 8;
  const int bk = (lane >> 4) * 8;

  // ---- load Q tile (64 x 128) ----
  {
    const int rr = tid >> 4, d8 = (tid & 15) * 8;
#pragma unroll
    for (int it = 0; it < 8; ++it) {
      const int r = rr + it * 8;
      const __bf16* src = Q + (size_t)(q0 + r) * 4096 + h * 128 + d8;
#if USE_ASYNC_LDS
      async_copy16(&Qs[r][d8], src);
#else
      *(v8bf*)&Qs[r][d8] = *(const v8bf*)src;
#endif
    }
  }

  v8f o[8];
#pragma unroll
  for (int n = 0; n < 8; ++n)
#pragma unroll
    for (int g = 0; g < 8; ++g) o[n][g] = 0.0f;
  float mrow[8], lrow[8];
#pragma unroll
  for (int g = 0; g < 8; ++g) { mrow[g] = -1.0e30f; lrow[g] = 0.0f; }

  for (int kc = 0; kc < 2048; kc += 64) {
    __syncthreads();
    // ---- stage K chunk (async rows) and V chunk (transposed) ----
    {
      const int rr = tid >> 4, d8 = (tid & 15) * 8;
#pragma unroll
      for (int it = 0; it < 8; ++it) {
        const int r = rr + it * 8;
        const size_t base = ((size_t)kvh * 2048 + kc + r) * 128 + d8;
#if USE_ASYNC_LDS
        async_copy16(&Ks[r][d8], Kc + base);
#else
        *(v8bf*)&Ks[r][d8] = *(const v8bf*)(Kc + base);
#endif
        v8bf vv = *(const v8bf*)(Vc + base);
#pragma unroll
        for (int j = 0; j < 8; ++j) Vt[d8 + j][r] = vv[j];
      }
    }
#if USE_ASYNC_LDS
    wait_async0();                       // async LDS writes landed
#endif
    __syncthreads();

    // ---- S = Q K^T : wave rows [w*16, w*16+16), 64 key columns ----
    v8f s[4];
#pragma unroll
    for (int j = 0; j < 4; ++j)
#pragma unroll
      for (int g = 0; g < 8; ++g) s[j][g] = 0.0f;

#pragma unroll
    for (int ks = 0; ks < 4; ++ks) {
      v16bf aq;
      {
        const int r = w * 16 + cl, k0 = ks * 32 + hl;
        v8bf lo = *(const v8bf*)&Qs[r][k0];
        v8bf hi = *(const v8bf*)&Qs[r][k0 + 16];
#pragma unroll
        for (int i = 0; i < 8; ++i) { aq[i] = lo[i]; aq[i + 8] = hi[i]; }
      }
#pragma unroll
      for (int j = 0; j < 4; ++j) {
        v16bf bkf;
        {
          const int n = j * 16 + cl, k0 = ks * 32 + bk;
          v8bf lo = *(const v8bf*)&Ks[n][k0];
          v8bf hi = *(const v8bf*)&Ks[n][k0 + 16];
#pragma unroll
          for (int i = 0; i < 8; ++i) { bkf[i] = lo[i]; bkf[i + 8] = hi[i]; }
        }
        s[j] = wmma_bf16(aq, bkf, s[j]);
      }
    }

    // ---- online softmax (C layout: element g of a half = one full row) ----
#pragma unroll
    for (int g = 0; g < 8; ++g) {
      float mc = fmaxf(fmaxf(s[0][g], s[1][g]), fmaxf(s[2][g], s[3][g]));
      mc = fmaxf(mc, __shfl_xor(mc, 1, 32));
      mc = fmaxf(mc, __shfl_xor(mc, 2, 32));
      mc = fmaxf(mc, __shfl_xor(mc, 4, 32));
      mc = fmaxf(mc, __shfl_xor(mc, 8, 32));
      mc *= scale;
      const float mn = fmaxf(mrow[g], mc);
      const float alpha = __expf(mrow[g] - mn);
      mrow[g] = mn;
      float psum = 0.0f;
      const int lr2 = g + hl;
#pragma unroll
      for (int j = 0; j < 4; ++j) {
        const float p = __expf(s[j][g] * scale - mn);
        psum += p;
        Ps[w][lr2][j * 16 + cl] = (__bf16)p;
      }
      psum += __shfl_xor(psum, 1, 32);
      psum += __shfl_xor(psum, 2, 32);
      psum += __shfl_xor(psum, 4, 32);
      psum += __shfl_xor(psum, 8, 32);
      lrow[g] = lrow[g] * alpha + psum;
#pragma unroll
      for (int n = 0; n < 8; ++n) o[n][g] *= alpha;
    }

    // ---- O += P(16x64) * V(64x128) ----
#pragma unroll
    for (int ks = 0; ks < 2; ++ks) {
      v16bf ap;
      {
        const int k0 = ks * 32 + hl;
        v8bf lo = *(const v8bf*)&Ps[w][cl][k0];
        v8bf hi = *(const v8bf*)&Ps[w][cl][k0 + 16];
#pragma unroll
        for (int i = 0; i < 8; ++i) { ap[i] = lo[i]; ap[i + 8] = hi[i]; }
      }
#pragma unroll
      for (int n = 0; n < 8; ++n) {
        v16bf bv;
        {
          const int d = n * 16 + cl, k0 = ks * 32 + bk;
          v8bf lo = *(const v8bf*)&Vt[d][k0];
          v8bf hi = *(const v8bf*)&Vt[d][k0 + 16];
#pragma unroll
          for (int i = 0; i < 8; ++i) { bv[i] = lo[i]; bv[i + 8] = hi[i]; }
        }
        o[n] = wmma_bf16(ap, bv, o[n]);
      }
    }
  }

  // ---- normalize and write attn output [s][h*128+d] ----
#pragma unroll
  for (int g = 0; g < 8; ++g) {
    const float rcp = 1.0f / lrow[g];
    const int r = q0 + w * 16 + g + hl;
#pragma unroll
    for (int n = 0; n < 8; ++n)
      Oa[(size_t)r * 4096 + h * 128 + n * 16 + cl] = (__bf16)(o[n][g] * rcp);
  }
}

// ---------------------------------------------------------------------------
extern "C" void kernel_launch(void* const* d_in, const int* in_sizes, int n_in,
                              void* d_out, int out_size, void* d_ws, size_t ws_size,
                              hipStream_t stream)
{
  const float* hidden = (const float*)d_in[0];
  const float* past_k = (const float*)d_in[1];
  const float* past_v = (const float*)d_in[2];
  const float* Wq     = (const float*)d_in[3];
  const float* Wk     = (const float*)d_in[4];
  const float* Wv     = (const float*)d_in[5];
  const float* Wo     = (const float*)d_in[6];
  float* out = (float*)d_out;

  char* w = (char*)d_ws;
  __bf16* Qbuf = (__bf16*)w; w += (size_t)1024 * 4096 * 2;    // Q projection (bf16)
  __bf16* Kp   = (__bf16*)w; w += (size_t)1024 * 1024 * 2;    // fresh K projection
  __bf16* Vp   = (__bf16*)w; w += (size_t)1024 * 1024 * 2;    // fresh V projection
  __bf16* Kc   = (__bf16*)w; w += (size_t)8 * 2048 * 128 * 2; // K cache (RoPE'd)
  __bf16* Vc   = (__bf16*)w; w += (size_t)8 * 2048 * 128 * 2; // V cache
  __bf16* Attn = (__bf16*)w; w += (size_t)1024 * 4096 * 2;    // attention output

  // Past KV -> bf16 cache with RoPE on keys.
  ingest_past_kernel<<<dim3(1024, 8), 64, 0, stream>>>(past_k, past_v, Kc, Vc);

  // Projections (fp32 in, bf16 out), all on WMMA.
  gemm_wmma<float, __bf16><<<dim3(32, 8), 256, 0, stream>>>(hidden, Wq, Qbuf, 1024, 4096, 4096);
  gemm_wmma<float, __bf16><<<dim3(8, 8),  256, 0, stream>>>(hidden, Wk, Kp,   1024, 1024, 4096);
  gemm_wmma<float, __bf16><<<dim3(8, 8),  256, 0, stream>>>(hidden, Wv, Vp,   1024, 1024, 4096);

  // RoPE fresh Q in place; RoPE + scatter fresh K, scatter fresh V.
  rope_q_kernel<<<dim3(1024, 32), 64, 0, stream>>>(Qbuf);
  scatter_new_kv_kernel<<<dim3(1024, 8), 64, 0, stream>>>(Kp, Vp, Kc, Vc);

  // Flash attention (WMMA for QK^T and PV, fp32 online softmax).
  flash_attn_kernel<<<dim3(16, 32), 128, 0, stream>>>(Qbuf, Kc, Vc, Attn);

  // Output projection (bf16 A, fp32 weights, fp32 out).
  gemm_wmma<__bf16, float><<<dim3(32, 8), 256, 0, stream>>>(Attn, Wo, out, 1024, 4096, 4096);
}
